// AttentionModel_39848706573278
// MI455X (gfx1250) — matrix operand order repaired and checked by reading, not compiled, not run on hardware
//
#include <hip/hip_runtime.h>
#include <hip/hip_bf16.h>

// ---------------------------------------------------------------------------
// MHA (proj Q/K/V + flash attention) for gfx1250 / MI455X.
//   B=4, S=2048, H=8, D=64.  f16 WMMA (16x16x32), f32 accumulate.
//   Kernel 1: X @ W^T + b  -> f16, Q scaled by D^-0.5, V stored transposed.
//   Kernel 2: flash attention; K/V blocks staged into LDS with
//             GLOBAL_LOAD_ASYNC_TO_LDS_B128 (ASYNCcnt), double buffered.
// ---------------------------------------------------------------------------

typedef __attribute__((ext_vector_type(16))) _Float16 v16h;
typedef __attribute__((ext_vector_type(8)))  _Float16 v8h;
typedef __attribute__((ext_vector_type(8)))  float    v8f;
typedef __attribute__((ext_vector_type(4)))  float    v4f;

#define B_SZ      4
#define S_LEN     2048
#define HEADS     8
#define DIM       64
#define INV_SCALE 0.125f

static __device__ __forceinline__ v8f wmma_f16(v16h a, v16h b, v8f c) {
  // D = A(16x32 f16) * B(32x16 f16) + C(16x16 f32)
  return __builtin_amdgcn_wmma_f32_16x16x32_f16(
      /*neg_a=*/false, a, /*neg_b=*/false, b,
      /*c_mod=*/(short)0, c, /*reuse_a=*/false, /*reuse_b=*/false);
}

// Wave-relative LDS byte address = low 32 bits of the flat (generic) address
// of an LDS object (ISA 10.2: LDS aperture keeps the offset in addr[31:0]).
static __device__ __forceinline__ unsigned lds_addr(const void* p) {
  return (unsigned)(unsigned long long)p;
}

// GLOBAL_LOAD_ASYNC_TO_LDS_B128, GVS mode: mem = saddr64 + vaddr32 (+0).
// vdst VGPR carries the LDS byte address. Tracked by ASYNCcnt.
static __device__ __forceinline__ void async_b128(unsigned lds_byte, unsigned goff,
                                                  unsigned long long sbase) {
  asm volatile("global_load_async_to_lds_b128 %0, %1, %2"
               :: "v"(lds_byte), "v"(goff), "s"(sbase) : "memory");
}

// Stage one 32-key block: K tile 32x64 halves (4KB, contiguous in [S,D]) and
// V^T tile 64x32 halves (64 rows x 64B, row stride S*2 in [D,S]).
// 128 threads x 2 chunks of 16B each per tensor -> 4 async ops per wave.
static __device__ __forceinline__ void issue_kv(const _Float16* Kb, const _Float16* Vb,
                                                int kb, unsigned klds, unsigned vlds,
                                                int tid) {
  const unsigned long long kbase = (unsigned long long)(Kb + (size_t)kb * DIM);
  const unsigned long long vbase = (unsigned long long)Vb;
#pragma unroll
  for (int i = 0; i < 2; ++i) {
    const unsigned c = (unsigned)tid + (unsigned)i * 128u;   // 16B chunk id
    async_b128(klds + c * 16u, c * 16u, kbase);
    const unsigned d = c >> 2, q4 = c & 3u;                  // V row / 16B col
    async_b128(vlds + c * 16u,
               d * (unsigned)(S_LEN * 2) + (unsigned)kb * 2u + q4 * 16u, vbase);
  }
}

// A-matrix 16x32 f16 layout: lane row m = lane&15; per lane the
// 16 halves are 16 contiguous K values [base0, base0+16)
// with base0 = (lane>>4)*16 (+32 for the second K-chunk).
static __device__ __forceinline__ v16h a_from_f32row(const float* row, int base0) {
  const v4f* p0 = (const v4f*)(row + base0);
  const v4f* p1 = (const v4f*)(row + base0 + 8);
  v4f f0 = p0[0], f1 = p0[1], f2 = p1[0], f3 = p1[1];
  v16h r;
#pragma unroll
  for (int i = 0; i < 4; ++i) {
    r[i]      = (_Float16)f0[i];
    r[4 + i]  = (_Float16)f1[i];
    r[8 + i]  = (_Float16)f2[i];
    r[12 + i] = (_Float16)f3[i];
  }
  return r;
}

// B-matrix 32x16 f16 layout: lane col n = lane&15, per lane 16 contiguous K
// values starting at (lane>>4)*16 (+32 for second K-chunk).
static __device__ __forceinline__ v16h b_from_f32(const float* p) {
  const v4f* q = (const v4f*)p;
  v4f f0 = q[0], f1 = q[1], f2 = q[2], f3 = q[3];
  v16h r;
#pragma unroll
  for (int i = 0; i < 4; ++i) {
    r[i]      = (_Float16)f0[i];
    r[4 + i]  = (_Float16)f1[i];
    r[8 + i]  = (_Float16)f2[i];
    r[12 + i] = (_Float16)f3[i];
  }
  return r;
}

static __device__ __forceinline__ v16h a_from_halves(const _Float16* row, int base0) {
  v8h g0 = *(const v8h*)(row + base0);
  v8h g1 = *(const v8h*)(row + base0 + 8);
  v16h r;
#pragma unroll
  for (int i = 0; i < 8; ++i) { r[i] = g0[i]; r[8 + i] = g1[i]; }
  return r;
}

// ---------------------------------------------------------------------------
// Kernel 1: projections. 4 waves/block, each wave -> 16 rows of 64 outputs
// for each of Q,K,V.  Rows are flat over [B,S,H].  Memory-bound.
// ---------------------------------------------------------------------------
__global__ void __launch_bounds__(128)
proj_kernel(const float* __restrict__ q_in, const float* __restrict__ k_in,
            const float* __restrict__ v_in,
            const float* __restrict__ Wq, const float* __restrict__ bq,
            const float* __restrict__ Wk, const float* __restrict__ bk,
            const float* __restrict__ Wv, const float* __restrict__ bv,
            _Float16* __restrict__ Qp, _Float16* __restrict__ Kp,
            _Float16* __restrict__ Vt) {
  const int lane = threadIdx.x & 31;
  const int wave = threadIdx.x >> 5;
  const int hi   = lane >> 4;
  const int lr   = lane & 15;
  const int r0   = blockIdx.x * 64 + wave * 16;

  const float* Xs[3] = {q_in, k_in, v_in};
  const float* Ws[3] = {Wq, Wk, Wv};
  const float* Bv[3] = {bq, bk, bv};

#pragma unroll
  for (int mat = 0; mat < 3; ++mat) {
    const float* xrow = Xs[mat] + (size_t)(r0 + lr) * DIM;
    v16h a0 = a_from_f32row(xrow, hi * 16);        // K-chunk d=0..31
    v16h a1 = a_from_f32row(xrow, hi * 16 + 32);   // K-chunk d=32..63

#pragma unroll
    for (int nt = 0; nt < 4; ++nt) {
      const int e = nt * 16 + lr;                 // output feature (B-col)
      v16h b0 = b_from_f32(Ws[mat] + e * DIM + hi * 16);
      v16h b1 = b_from_f32(Ws[mat] + e * DIM + hi * 16 + 32);
      v8f c = {};
      c = wmma_f16(a1, b1, c);
      c = wmma_f16(a0, b0, c);
      const float bias = Bv[mat][e];

#pragma unroll
      for (int j = 0; j < 8; ++j) {               // C rows: m = hi*8+j
        const int r   = r0 + hi * 8 + j;
        const int b   = r >> 14;                  // / (S*H) = 16384
        const int rem = r & 16383;
        const int s   = rem >> 3;                 // / HEADS
        const int h   = rem & 7;
        const int bh  = (b << 3) + h;
        float val = c[j] + bias;
        if (mat == 0) val *= INV_SCALE;           // fold softmax scale into Q
        const _Float16 hv = (_Float16)val;
        if (mat == 0)      Qp[((size_t)bh * S_LEN + s) * DIM + e] = hv;
        else if (mat == 1) Kp[((size_t)bh * S_LEN + s) * DIM + e] = hv;
        else               Vt[((size_t)bh * DIM + e) * S_LEN + s] = hv; // V^T
      }
    }
  }
}

// ---------------------------------------------------------------------------
// Kernel 2: flash attention. grid = (S/64, B*H), 4 waves/block,
// wave owns 16 queries; block streams one 32-key K/V tile per step through
// async-DMA'd, double-buffered LDS shared by all 4 waves.
// ---------------------------------------------------------------------------
__global__ void __launch_bounds__(128)
attn_kernel(const _Float16* __restrict__ Qp, const _Float16* __restrict__ Kp,
            const _Float16* __restrict__ Vt, float* __restrict__ out) {
  __shared__ alignas(64) _Float16 Klds[2][32][64];   // [buf][key][d]      8 KB
  __shared__ alignas(64) _Float16 Vlds[2][64][32];   // [buf][d][key]      8 KB
  __shared__ alignas(64) _Float16 Plds[4][16][32];   // per-wave P stage   4 KB

  const int tid  = threadIdx.x;
  const int lane = tid & 31;
  const int wave = tid >> 5;
  const int hi   = lane >> 4;
  const int lr   = lane & 15;
  const int bh   = blockIdx.y;
  const int q0   = blockIdx.x * 64 + wave * 16;

  const _Float16* Qb = Qp + (size_t)bh * S_LEN * DIM;
  const _Float16* Kb = Kp + (size_t)bh * S_LEN * DIM;
  const _Float16* Vb = Vt + (size_t)bh * DIM * S_LEN;

  const unsigned klds0 = lds_addr(&Klds[0][0][0]);
  const unsigned klds1 = lds_addr(&Klds[1][0][0]);
  const unsigned vlds0 = lds_addr(&Vlds[0][0][0]);
  const unsigned vlds1 = lds_addr(&Vlds[1][0][0]);

  // Prologue: start DMA of the first K/V block into buffer 0.
  issue_kv(Kb, Vb, 0, klds0, vlds0, tid);

  // Q A-tiles held in registers for the whole kernel (Q pre-scaled).
  const _Float16* qrow = Qb + (size_t)(q0 + lr) * DIM;
  const v16h qa0 = a_from_halves(qrow, hi * 16);
  const v16h qa1 = a_from_halves(qrow, hi * 16 + 32);

  float m_run[8], l_run[8];
  v8f acc[4] = {v8f{}, v8f{}, v8f{}, v8f{}};
#pragma unroll
  for (int j = 0; j < 8; ++j) { m_run[j] = -__builtin_inff(); l_run[j] = 0.0f; }

  for (int kb = 0; kb < S_LEN; kb += 32) {
    const int buf = (kb >> 5) & 1;

    // Kick the DMA for the next block, then wait only for the current one:
    // async ops complete in order, so asynccnt<=4 ==> the older 4 are done.
    if (kb + 32 < S_LEN) {
      issue_kv(Kb, Vb, kb + 32, buf ? klds0 : klds1, buf ? vlds0 : vlds1, tid);
      asm volatile("s_wait_asynccnt 4" ::: "memory");
    } else {
      asm volatile("s_wait_asynccnt 0" ::: "memory");
    }
    __syncthreads();   // all waves' portions of the current buffer resident

    // ---- scores: two 16-key groups, K=64 contraction in 2 WMMA each ----
    v8f s0 = {}, s1 = {};
    {
      const _Float16* kr = &Klds[buf][lr][hi * 16];
      s0 = wmma_f16(qa1, *(const v16h*)(kr + 32), s0);
      s0 = wmma_f16(qa0, *(const v16h*)(kr), s0);
    }
    {
      const _Float16* kr = &Klds[buf][16 + lr][hi * 16];
      s1 = wmma_f16(qa1, *(const v16h*)(kr + 32), s1);
      s1 = wmma_f16(qa0, *(const v16h*)(kr), s1);
    }

    // ---- online softmax (row reductions across the 16-lane C groups) ----
    float scale_[8];
#pragma unroll
    for (int j = 0; j < 8; ++j) {
      float t = fmaxf(s0[j], s1[j]);
      t = fmaxf(t, __shfl_xor(t, 1, 32));
      t = fmaxf(t, __shfl_xor(t, 2, 32));
      t = fmaxf(t, __shfl_xor(t, 4, 32));
      t = fmaxf(t, __shfl_xor(t, 8, 32));
      const float mn = fmaxf(m_run[j], t);
      const float sc = __expf(m_run[j] - mn);
      const float p0 = __expf(s0[j] - mn);
      const float p1 = __expf(s1[j] - mn);
      float rs = p0 + p1;
      rs += __shfl_xor(rs, 1, 32);
      rs += __shfl_xor(rs, 2, 32);
      rs += __shfl_xor(rs, 4, 32);
      rs += __shfl_xor(rs, 8, 32);
      l_run[j] = l_run[j] * sc + rs;
      m_run[j] = mn;
      scale_[j] = sc;
      s0[j] = p0;
      s1[j] = p1;
    }
#pragma unroll
    for (int dt = 0; dt < 4; ++dt)
#pragma unroll
      for (int j = 0; j < 8; ++j) acc[dt][j] *= scale_[j];

    // ---- transpose P (C-layout -> A-layout) via per-wave LDS region ----
#pragma unroll
    for (int j = 0; j < 8; ++j) {
      const int m = hi * 8 + j;
      Plds[wave][m][lr]      = (_Float16)s0[j];
      Plds[wave][m][16 + lr] = (_Float16)s1[j];
    }
    asm volatile("s_wait_dscnt 0" ::: "memory");  // own-wave DS ops only
    v16h pa;
    {
      v8h g0 = *(const v8h*)&Plds[wave][lr][hi * 16];
      v8h g1 = *(const v8h*)&Plds[wave][lr][hi * 16 + 8];
#pragma unroll
      for (int i = 0; i < 8; ++i) { pa[i] = g0[i]; pa[8 + i] = g1[i]; }
    }

    // ---- acc += P (16x32) * V (32x16) per 16-wide dim tile ----
#pragma unroll
    for (int dt = 0; dt < 4; ++dt) {
      const _Float16* vr = &Vlds[buf][dt * 16 + lr][hi * 16];
      acc[dt] = wmma_f16(pa, *(const v16h*)vr, acc[dt]);
    }

    __syncthreads();   // everyone done with this buffer before it is re-filled
  }

  // ---- epilogue: normalize and scatter to [B,S,H,D] f32 ----
  const int b = bh >> 3;
  const int h = bh & 7;
#pragma unroll
  for (int j = 0; j < 8; ++j) {
    const float inv = 1.0f / l_run[j];
    const int srow = q0 + hi * 8 + j;
    const size_t base = (((size_t)b * S_LEN + srow) * HEADS + h) * DIM;
#pragma unroll
    for (int dt = 0; dt < 4; ++dt)
      out[base + dt * 16 + lr] = acc[dt][j] * inv;
  }
}

// ---------------------------------------------------------------------------
extern "C" void kernel_launch(void* const* d_in, const int* in_sizes, int n_in,
                              void* d_out, int out_size, void* d_ws, size_t ws_size,
                              hipStream_t stream) {
  (void)in_sizes; (void)n_in; (void)out_size; (void)ws_size;
  const float* q  = (const float*)d_in[0];
  const float* k  = (const float*)d_in[1];
  const float* v  = (const float*)d_in[2];
  const float* Wq = (const float*)d_in[3];
  const float* bq = (const float*)d_in[4];
  const float* Wk = (const float*)d_in[5];
  const float* bk = (const float*)d_in[6];
  const float* Wv = (const float*)d_in[7];
  const float* bv = (const float*)d_in[8];
  float* out = (float*)d_out;

  const size_t per = (size_t)B_SZ * HEADS * S_LEN * DIM;  // 4Mi halves = 8 MB
  _Float16* Qp = (_Float16*)d_ws;
  _Float16* Kp = Qp + per;
  _Float16* Vt = Kp + per;                                // total ws: 24 MB

  proj_kernel<<<dim3((B_SZ * S_LEN * HEADS) / 64), dim3(128), 0, stream>>>(
      q, k, v, Wq, bq, Wk, bk, Wv, bv, Qp, Kp, Vt);
  attn_kernel<<<dim3(S_LEN / 64, B_SZ * HEADS), dim3(128), 0, stream>>>(
      Qp, Kp, Vt, out);
}
